// CausalSelfAttention_24386824307240
// MI455X (gfx1250) — matrix operand order repaired and checked by reading, not compiled, not run on hardware
//
#include <hip/hip_runtime.h>
#include <hip/hip_bf16.h>

// ---------------- problem constants ----------------
#define kB   2
#define kT   2048
#define kC   1024
#define kH   16
#define kHD  64
#define k3C  (3 * kC)
#define kM   (kB * kT)          // 4096 rows of x

// q pre-scale: (1/sqrt(HD)) * log2(e), so softmax can use native base-2 exp
#define kQSCALE 0.18033688011112042f

typedef __bf16 bf16;
typedef bf16  v16bf __attribute__((ext_vector_type(16)));
typedef bf16  v8bf  __attribute__((ext_vector_type(8)));
typedef float v8f   __attribute__((ext_vector_type(8)));

static __device__ inline v8f zero8() {
    v8f z;
    #pragma unroll
    for (int i = 0; i < 8; ++i) z[i] = 0.0f;
    return z;
}

static __device__ inline v8f wmma_bf16(v16bf a, v16bf b, v8f c) {
    // D = A(16x32) * B(32x16) + C(16x16), fp32 accumulate
    return __builtin_amdgcn_wmma_f32_16x16x32_bf16(
        /*neg_a=*/false, a, /*neg_b=*/false, b,
        /*c_mod=*/(short)0, c, /*reuse_a=*/false, /*reuse_b=*/false);
}

// A fragment: 16x32 bf16. lane<16: M=lane, K = {k0+0..7, k0+16..23}
//             lane>=16: M=lane-16, K = {k0+8..15, k0+24..31}
static __device__ inline v16bf load_a_frag(const bf16* __restrict__ base,
                                           int ld, int row0, int k0) {
    int lane = threadIdx.x & 31;
    int r  = row0 + (lane & 15);
    int hi = lane >> 4;
    const bf16* p = base + (size_t)r * ld + k0 + hi * 8;
    union { v16bf f; v8bf h[2]; } u;
    u.h[0] = *(const v8bf*)p;
    u.h[1] = *(const v8bf*)(p + 16);
    return u.f;
}

// B fragment: 32x16 bf16 read from row-major [N,K] source (per-lane row = n).
// lane<16: N=lane, K = {k0+0..7, k0+16..23} ; lane>=16: N=lane-16, K = {k0+8..15, k0+24..31}
static __device__ inline v16bf load_b_frag(const bf16* __restrict__ base,
                                           int ld, int n0, int k0) {
    int lane = threadIdx.x & 31;
    int r  = n0 + (lane & 15);
    int hi = lane >> 4;
    const bf16* p = base + (size_t)r * ld + k0 + hi * 8;
    union { v16bf f; v8bf h[2]; } u;
    u.h[0] = *(const v8bf*)p;
    u.h[1] = *(const v8bf*)(p + 16);
    return u.f;
}

// ---------------- fp32 -> bf16 convert ----------------
__global__ void cvt_f32_bf16_kernel(const float* __restrict__ in,
                                    bf16* __restrict__ out, int n) {
    int i = blockIdx.x * blockDim.x + threadIdx.x;
    int stride = gridDim.x * blockDim.x;
    for (; i < n; i += stride) out[i] = (bf16)in[i];
}

// ---------------- GEMM: out = A(bf16 [M,K]) * W(bf16 [N,K])^T + bias ------
// Each wave: 64x64 output tile (16 WMMA per K-step). Block: 8 waves = 256x128.
// MODE 0: float output (projection).
// MODE 1: bf16 output (qkv); scale q (col<kC) by kQSCALE; also scatter a
//         transposed copy of V into VT[B,H,HD,T] for contiguous PV fragments.
template <int MODE>
__global__ __launch_bounds__(256)
void gemm_wmma_kernel(const bf16* __restrict__ A, const bf16* __restrict__ W,
                      const float* __restrict__ bias, void* __restrict__ Out,
                      bf16* __restrict__ VT, int Mdim, int Ndim, int Kdim) {
    int wave = threadIdx.x >> 5;
    int lane = threadIdx.x & 31;
    int m0 = blockIdx.x * 256 + (wave & 3) * 64;   // 4 wave-rows of 64
    int n0 = blockIdx.y * 128 + (wave >> 2) * 64;  // 2 wave-cols of 64

    v8f acc[4][4];
    #pragma unroll
    for (int mi = 0; mi < 4; ++mi)
        #pragma unroll
        for (int ni = 0; ni < 4; ++ni) acc[mi][ni] = zero8();

    for (int k0 = 0; k0 < Kdim; k0 += 32) {
        v16bf af[4], bf_[4];
        #pragma unroll
        for (int i = 0; i < 4; ++i) af[i]  = load_a_frag(A, Kdim, m0 + 16 * i, k0);
        #pragma unroll
        for (int i = 0; i < 4; ++i) bf_[i] = load_b_frag(W, Kdim, n0 + 16 * i, k0);
        #pragma unroll
        for (int mi = 0; mi < 4; ++mi)
            #pragma unroll
            for (int ni = 0; ni < 4; ++ni)
                acc[mi][ni] = wmma_bf16(af[mi], bf_[ni], acc[mi][ni]);
    }

    #pragma unroll
    for (int mi = 0; mi < 4; ++mi) {
        #pragma unroll
        for (int ni = 0; ni < 4; ++ni) {
            int col = n0 + ni * 16 + (lane & 15);
            float bv = bias[col];
            #pragma unroll
            for (int j = 0; j < 8; ++j) {
                int row = m0 + mi * 16 + j + (lane >> 4) * 8;
                float val = acc[mi][ni][j] + bv;
                if (MODE == 0) {
                    ((float*)Out)[(size_t)row * Ndim + col] = val;
                } else {
                    float sv = (col < kC) ? val * kQSCALE : val;
                    ((bf16*)Out)[(size_t)row * Ndim + col] = (bf16)sv;
                    if (col >= 2 * kC) {
                        int f = col - 2 * kC;            // h*HD + d
                        int b = row / kT, t = row % kT;
                        int h = f >> 6, d = f & 63;
                        VT[((size_t)(b * kH + h) * kHD + d) * kT + t] = (bf16)val;
                    }
                }
            }
        }
    }
}

// ---- one 16(q) x 64(kv) flash-attention slab: 8 WMMA S + 8 WMMA PV -------
// MASKED=false for slabs fully below the causal diagonal (no cmp/cndmask).
// Row-sum kept as per-lane partials (lpart); reduced once at kernel exit.
template <bool MASKED>
static __device__ inline void attn_slab(int kv0, int q0, int lane,
                                        const bf16* __restrict__ kbase,
                                        const bf16* __restrict__ vbase,
                                        bf16* pl,
                                        const v16bf& aq0, const v16bf& aq1,
                                        v8f o[4], float mrow[8], float lpart[8]) {
    v8f s[4];
    #pragma unroll
    for (int nt = 0; nt < 4; ++nt) {
        int n0 = kv0 + nt * 16;
        v16bf bk0 = load_b_frag(kbase, k3C, n0, 0);
        v16bf bk1 = load_b_frag(kbase, k3C, n0, 32);
        v8f t = wmma_bf16(aq0, bk0, zero8());
        s[nt]   = wmma_bf16(aq1, bk1, t);
        if (MASKED) {
            int colg = n0 + (lane & 15);
            #pragma unroll
            for (int j = 0; j < 8; ++j) {
                int row = q0 + j + (lane >> 4) * 8;
                if (colg > row) s[nt][j] = -1e30f;   // causal mask
            }
        }
    }
    // online softmax over this 16x64 slab (stats per C-layout row, base-2)
    #pragma unroll
    for (int j = 0; j < 8; ++j) {
        float v = fmaxf(fmaxf(s[0][j], s[1][j]), fmaxf(s[2][j], s[3][j]));
        #pragma unroll
        for (int off = 8; off; off >>= 1) v = fmaxf(v, __shfl_xor(v, off, 32));
        float mnew  = fmaxf(mrow[j], v);
        float scale = exp2f(mrow[j] - mnew);
        mrow[j] = mnew;
        float p0 = exp2f(s[0][j] - mnew);
        float p1 = exp2f(s[1][j] - mnew);
        float p2 = exp2f(s[2][j] - mnew);
        float p3 = exp2f(s[3][j] - mnew);
        lpart[j] = lpart[j] * scale + ((p0 + p1) + (p2 + p3));
        #pragma unroll
        for (int t2 = 0; t2 < 4; ++t2) o[t2][j] *= scale;
        // stash P (bf16) row-major [16][64] in this wave's LDS slab
        int rl = j + (lane >> 4) * 8;
        pl[rl * 64 +      (lane & 15)] = (bf16)p0;
        pl[rl * 64 + 16 + (lane & 15)] = (bf16)p1;
        pl[rl * 64 + 32 + (lane & 15)] = (bf16)p2;
        pl[rl * 64 + 48 + (lane & 15)] = (bf16)p3;
    }
    // same-wave LDS in-order: wait for stores, then re-read in A-layout
    asm volatile("s_wait_dscnt 0" ::: "memory");
    v16bf ap0, ap1;
    {
        int r = lane & 15, hi = lane >> 4;
        const bf16* pp = pl + r * 64 + hi * 8;
        union { v16bf f; v8bf h2[2]; } u;
        u.h2[0] = *(const v8bf*)pp;
        u.h2[1] = *(const v8bf*)(pp + 16);
        ap0 = u.f;
        u.h2[0] = *(const v8bf*)(pp + 32);
        u.h2[1] = *(const v8bf*)(pp + 48);
        ap1 = u.f;
    }
    // O(16x64) += P(16x64) @ V(64x64), via transposed-V fragments
    #pragma unroll
    for (int t2 = 0; t2 < 4; ++t2) {
        o[t2] = wmma_bf16(ap0, load_b_frag(vbase, kT, t2 * 16, kv0),      o[t2]);
        o[t2] = wmma_bf16(ap1, load_b_frag(vbase, kT, t2 * 16, kv0 + 32), o[t2]);
    }
}

// ---------------- flash attention: one wave = (b, h, 16 query rows) -------
__global__ __launch_bounds__(256)
void attn_wmma_kernel(const bf16* __restrict__ qkv,  // [B*T, 3C] (q pre-scaled by kQSCALE)
                      const bf16* __restrict__ vt,   // [B,H,HD,T]
                      bf16* __restrict__ y) {        // [B*T, C]
    __shared__ __attribute__((aligned(32))) bf16 plds[8][16 * 64];

    int wave = threadIdx.x >> 5;
    int lane = threadIdx.x & 31;
    int w  = blockIdx.x * 8 + wave;
    const int QT = kT / 16;                 // 128 q-tiles per (b,h)
    int qt = w % QT;
    int bh = w / QT;
    int b = bh / kH, h = bh % kH;
    int q0 = qt * 16;

    const bf16* qbase = qkv + (size_t)b * kT * k3C +       (size_t)h * kHD;
    const bf16* kbase = qkv + (size_t)b * kT * k3C + kC +  (size_t)h * kHD;
    const bf16* vbase = vt  + (size_t)(b * kH + h) * kHD * kT;  // [HD][T]

    v16bf aq0 = load_a_frag(qbase, k3C, q0, 0);
    v16bf aq1 = load_a_frag(qbase, k3C, q0, 32);

    v8f o[4];
    #pragma unroll
    for (int i = 0; i < 4; ++i) o[i] = zero8();
    float mrow[8], lpart[8];
    #pragma unroll
    for (int j = 0; j < 8; ++j) { mrow[j] = -1e30f; lpart[j] = 0.0f; }

    bf16* pl = plds[wave];
    int jt_max   = (q0 + 15) / 64;                          // last slab (kv0 <= q0)
    int n_unmask = (q0 >= 63) ? ((q0 - 63) / 64 + 1) : 0;   // slabs fully below diag

    for (int jt = 0; jt < n_unmask; ++jt)
        attn_slab<false>(jt * 64, q0, lane, kbase, vbase, pl, aq0, aq1, o, mrow, lpart);
    for (int jt = n_unmask; jt <= jt_max; ++jt)
        attn_slab<true>(jt * 64, q0, lane, kbase, vbase, pl, aq0, aq1, o, mrow, lpart);

    // final cross-lane row-sum reduction, then normalize + write y (bf16)
    #pragma unroll
    for (int j = 0; j < 8; ++j) {
        float lsum = lpart[j];
        #pragma unroll
        for (int off = 8; off; off >>= 1) lsum += __shfl_xor(lsum, off, 32);
        float inv = 1.0f / lsum;
        int row = q0 + j + (lane >> 4) * 8;
        size_t rbase = ((size_t)(b * kT + row)) * kC + h * kHD;
        #pragma unroll
        for (int t2 = 0; t2 < 4; ++t2)
            y[rbase + t2 * 16 + (lane & 15)] = (bf16)(o[t2][j] * inv);
    }
}

// ---------------- host-side orchestration ----------------
extern "C" void kernel_launch(void* const* d_in, const int* in_sizes, int n_in,
                              void* d_out, int out_size, void* d_ws, size_t ws_size,
                              hipStream_t stream) {
    const float* x      = (const float*)d_in[0];   // [B,T,C]
    const float* w_attn = (const float*)d_in[1];   // [3C,C]
    const float* b_attn = (const float*)d_in[2];   // [3C]
    const float* w_proj = (const float*)d_in[3];   // [C,C]
    const float* b_proj = (const float*)d_in[4];   // [C]

    char* ws = (char*)d_ws;
    size_t off = 0;
    bf16* xb   = (bf16*)(ws + off); off += (size_t)kM  * kC  * sizeof(bf16); // 8 MiB
    bf16* wab  = (bf16*)(ws + off); off += (size_t)k3C * kC  * sizeof(bf16); // 6 MiB
    bf16* wpb  = (bf16*)(ws + off); off += (size_t)kC  * kC  * sizeof(bf16); // 2 MiB
    bf16* qkvb = (bf16*)(ws + off); off += (size_t)kM  * k3C * sizeof(bf16); // 24 MiB
    bf16* vtb  = (bf16*)(ws + off); off += (size_t)kM  * kC  * sizeof(bf16); // 8 MiB
    bf16* yb   = (bf16*)(ws + off); off += (size_t)kM  * kC  * sizeof(bf16); // 8 MiB
    (void)ws_size; (void)in_sizes; (void)n_in; (void)out_size;

    // 1) downcast inputs to bf16
    {
        int n0 = kM * kC, n1 = k3C * kC, n2 = kC * kC;
        cvt_f32_bf16_kernel<<<(n0 + 255) / 256, 256, 0, stream>>>(x,      xb,  n0);
        cvt_f32_bf16_kernel<<<(n1 + 255) / 256, 256, 0, stream>>>(w_attn, wab, n1);
        cvt_f32_bf16_kernel<<<(n2 + 255) / 256, 256, 0, stream>>>(w_proj, wpb, n2);
    }
    // 2) QKV projection (q pre-scaled by 0.125*log2e, V^T materialized)
    gemm_wmma_kernel<1><<<dim3(kM / 256, k3C / 128), 256, 0, stream>>>(
        xb, wab, b_attn, (void*)qkvb, vtb, kM, k3C, kC);
    // 3) causal flash attention (4096 waves / 512 blocks)
    attn_wmma_kernel<<<(kB * kH * (kT / 16)) / 8, 256, 0, stream>>>(qkvb, vtb, yb);
    // 4) output projection -> fp32 result
    gemm_wmma_kernel<0><<<dim3(kM / 256, kC / 128), 256, 0, stream>>>(
        yb, wpb, b_proj, (void*)d_out, nullptr, kM, kC, kC);
}
